// DoubleStreamBlock_49185965474066
// MI455X (gfx1250) — matrix operand order repaired and checked by reading, not compiled, not run on hardware
//
#include <hip/hip_runtime.h>
#include <stdint.h>
#include <stddef.h>

#define HIDDEN 3072
#define MLPD   12288
#define NHEAD  24
#define HDIM   128
#define TIMG   1536
#define TTXT   512
#define LSEQ   2048

typedef __attribute__((ext_vector_type(16))) __bf16 v16bf;
typedef __attribute__((ext_vector_type(4)))  __bf16 v4bf;
typedef __attribute__((ext_vector_type(8)))  float  v8f;
typedef __attribute__((ext_vector_type(4)))  float  v4f;

// POD 16-byte vector so it can live inside a union (HIP uint4 has ctors).
struct alignas(16) U4 { unsigned int w[4]; };

union Frag16 {
    v16bf v;
    v4bf  b4[4];
    unsigned short s[16];
    U4 q[2];
};

// Native conversion: lowers to v_cvt_pk_bf16_f32 on CDNA5.
__device__ __forceinline__ unsigned short f2bf(float f) {
    return __builtin_bit_cast(unsigned short, (__bf16)f);
}

__device__ __forceinline__ v8f wmma_bf16(const Frag16& a, const Frag16& b, v8f c) {
    // (neg_a, A, neg_b, B, c_mod, C, reuse_a, reuse_b)
    return __builtin_amdgcn_wmma_f32_16x16x32_bf16(false, a.v, false, b.v,
                                                   (short)0, c, false, false);
}

// ---- fragment loaders (contiguous K per lane half, wave32) -------------------------
// A 16x32 bf16: lane 0-15 -> row lane, K = kb..kb+15 (kb=0);
//               lane 16-31 -> same rows, kb=16.
__device__ __forceinline__ Frag16 loadA(const unsigned short* base, int row0,
                                        int ld, int k0, int lane) {
    Frag16 f;
    const unsigned short* p = base + (size_t)(row0 + (lane & 15)) * ld
                                   + (k0 + ((lane >> 4) << 4));
    f.q[0] = *(const U4*)(p);
    f.q[1] = *(const U4*)(p + 8);
    return f;
}

// B 32x16 from f32 weights [N,K] row-major, converted to bf16 in-register
// with packed fptrunc (v_cvt_pk_bf16_f32).
__device__ __forceinline__ Frag16 loadB_f32(const float* base, int n0,
                                            int ld, int k0, int lane) {
    Frag16 f;
    const float* p = base + (size_t)(n0 + (lane & 15)) * ld
                          + (k0 + ((lane >> 4) << 4));
    f.b4[0] = __builtin_convertvector(*(const v4f*)(p),      v4bf);
    f.b4[1] = __builtin_convertvector(*(const v4f*)(p + 4),  v4bf);
    f.b4[2] = __builtin_convertvector(*(const v4f*)(p + 8),  v4bf);
    f.b4[3] = __builtin_convertvector(*(const v4f*)(p + 12), v4bf);
    return f;
}

// ---- LayerNorm + modulation: out = bf16(shift + (1+scale)*LN(x)) -----------
__global__ __launch_bounds__(256)
void ln_mod_kernel(const float* __restrict__ x, const float* __restrict__ mod3,
                   unsigned short* __restrict__ out) {
    const float* shift = mod3;
    const float* scale = mod3 + HIDDEN;
    const int t = blockIdx.x;
    const float* xr = x + (size_t)t * HIDDEN;
    float vals[12];
    float s = 0.f, q = 0.f;
#pragma unroll
    for (int i = 0; i < 12; ++i) {
        float v = xr[threadIdx.x + i * 256];
        vals[i] = v; s += v; q += v * v;
    }
    for (int m = 1; m < 32; m <<= 1) { s += __shfl_xor(s, m); q += __shfl_xor(q, m); }
    __shared__ float ss[8], sq[8];
    const int wave = threadIdx.x >> 5, lane = threadIdx.x & 31;
    if (lane == 0) { ss[wave] = s; sq[wave] = q; }
    __syncthreads();
    float S = 0.f, Q = 0.f;
#pragma unroll
    for (int i = 0; i < 8; ++i) { S += ss[i]; Q += sq[i]; }
    const float mean = S * (1.0f / HIDDEN);
    const float var  = Q * (1.0f / HIDDEN) - mean * mean;
    const float rs   = rsqrtf(var + 1e-6f);
    unsigned short* orow = out + (size_t)t * HIDDEN;
#pragma unroll
    for (int i = 0; i < 12; ++i) {
        int c = threadIdx.x + i * 256;
        orow[c] = f2bf(shift[c] + (1.0f + scale[c]) * (vals[i] - mean) * rs);
    }
}

// ---- generic WMMA GEMM: out = epilogue(A[M,K](bf16) @ W[N,K]^T(f32) + bias)
// EPI 0: f32 out.  1: GELU(tanh)->bf16.  2: res + gate*val -> f32.
// 3: clip(res + gate*val) -> f32.
template <int EPI>
__global__ __launch_bounds__(256)
void gemm_kernel(const unsigned short* __restrict__ A,
                 const float* __restrict__ W,
                 const float* __restrict__ bias,
                 float* __restrict__ outF,
                 unsigned short* __restrict__ outH,
                 const float* __restrict__ res,
                 const float* __restrict__ gate,
                 int M, int N, int K) {
    const int lane = threadIdx.x & 31;
    const int wave = threadIdx.x >> 5;
    const int m0 = blockIdx.x * 128 + (wave >> 2) * 64;
    const int n0 = blockIdx.y * 256 + (wave & 3) * 64;
    (void)M;

    v8f vzero;
#pragma unroll
    for (int i = 0; i < 8; ++i) vzero[i] = 0.0f;
    v8f acc[4][4];
#pragma unroll
    for (int i = 0; i < 4; ++i)
#pragma unroll
        for (int j = 0; j < 4; ++j) acc[i][j] = vzero;

    for (int k0 = 0; k0 < K; k0 += 32) {
        Frag16 af[4];
#pragma unroll
        for (int mi = 0; mi < 4; ++mi) af[mi] = loadA(A, m0 + mi * 16, K, k0, lane);
#pragma unroll
        for (int ni = 0; ni < 4; ++ni) {
            Frag16 bf = loadB_f32(W, n0 + ni * 16, K, k0, lane);
#pragma unroll
            for (int mi = 0; mi < 4; ++mi)
                acc[mi][ni] = wmma_bf16(af[mi], bf, acc[mi][ni]);
        }
    }

    const int rofs = (lane >> 4) << 3;
    const int cofs = lane & 15;
#pragma unroll
    for (int mi = 0; mi < 4; ++mi)
#pragma unroll
    for (int ni = 0; ni < 4; ++ni)
#pragma unroll
    for (int v = 0; v < 8; ++v) {
        const int row = m0 + mi * 16 + v + rofs;
        const int col = n0 + ni * 16 + cofs;
        const size_t o = (size_t)row * N + col;
        float val = acc[mi][ni][v] + bias[col];
        if (EPI == 0) {
            outF[o] = val;
        } else if (EPI == 1) {
            float t = val + 0.044715f * val * val * val;
            float g = 0.5f * val * (1.0f + tanhf(0.7978845608028654f * t));
            outH[o] = f2bf(g);
        } else {
            float r = res[o] + gate[col] * val;
            if (EPI == 3) r = fminf(fmaxf(r, -65504.0f), 65504.0f);
            outF[o] = r;
        }
    }
}

// ---- QKV post: RMS-norm(q,k) + RoPE, write Q/K [h][s][128], V^T [h][d][s] --
__global__ __launch_bounds__(128)
void qkv_post_kernel(const float* __restrict__ qkv,
                     const float* __restrict__ qscale,
                     const float* __restrict__ kscale,
                     const float* __restrict__ pe,   // [L,64,2,2]
                     unsigned short* __restrict__ Qb,
                     unsigned short* __restrict__ Kb,
                     unsigned short* __restrict__ Vt,
                     int seq_off) {
    const int t = blockIdx.x, h = blockIdx.y, d = threadIdx.x;
    const int s = seq_off + t;
    const float* row = qkv + (size_t)t * (3 * HIDDEN);
    float qv = row[h * HDIM + d];
    float kv = row[HIDDEN + h * HDIM + d];
    float vv = row[2 * HIDDEN + h * HDIM + d];

    float qs = qv * qv, ks = kv * kv;
    for (int m = 1; m < 32; m <<= 1) { qs += __shfl_xor(qs, m); ks += __shfl_xor(ks, m); }
    __shared__ float sh[8];
    const int wave = threadIdx.x >> 5, lane = threadIdx.x & 31;
    if (lane == 0) { sh[wave] = qs; sh[4 + wave] = ks; }
    __syncthreads();
    const float QS = sh[0] + sh[1] + sh[2] + sh[3];
    const float KS = sh[4] + sh[5] + sh[6] + sh[7];
    const float qr = rsqrtf(QS * (1.0f / HDIM) + 1e-6f);
    const float kr = rsqrtf(KS * (1.0f / HDIM) + 1e-6f);

    __shared__ float qn[HDIM], kn[HDIM];
    qn[d] = qv * qr * qscale[d];
    kn[d] = kv * kr * kscale[d];
    __syncthreads();

    const int p = d >> 1, i = d & 1;
    const float* pp = pe + (((size_t)s * 64 + p) * 2 + i) * 2;
    const float qo = pp[0] * qn[2 * p] + pp[1] * qn[2 * p + 1];
    const float ko = pp[0] * kn[2 * p] + pp[1] * kn[2 * p + 1];

    const size_t qidx = ((size_t)h * LSEQ + s) * HDIM + d;
    Qb[qidx] = f2bf(qo);
    Kb[qidx] = f2bf(ko);
    Vt[((size_t)h * HDIM + d) * LSEQ + s] = f2bf(vv);
}

// ---- flash attention: one wave per 16 query rows, online softmax -----------
__global__ __launch_bounds__(256)
void flash_kernel(const unsigned short* __restrict__ Qb,
                  const unsigned short* __restrict__ Kb,
                  const unsigned short* __restrict__ Vt,
                  unsigned short* __restrict__ O) {
    const int lane = threadIdx.x & 31;
    const int wave = threadIdx.x >> 5;
    const int h = blockIdx.y;
    const int qb = blockIdx.x * 8 + wave;
    const unsigned short* Qh = Qb + (size_t)h * LSEQ * HDIM;
    const unsigned short* Kh = Kb + (size_t)h * LSEQ * HDIM;
    const unsigned short* Vh = Vt + (size_t)h * HDIM * LSEQ;

    Frag16 qf[4];
#pragma unroll
    for (int j = 0; j < 4; ++j) qf[j] = loadA(Qh, qb * 16, HDIM, j * 32, lane);

    v8f vzero;
#pragma unroll
    for (int i = 0; i < 8; ++i) vzero[i] = 0.0f;
    v8f o[8];
#pragma unroll
    for (int i = 0; i < 8; ++i) o[i] = vzero;
    float mrow[8], lrow[8];
#pragma unroll
    for (int i = 0; i < 8; ++i) { mrow[i] = -3.0e38f; lrow[i] = 0.0f; }

    __shared__ __align__(16) unsigned short psh[8][16 * 32];   // per-wave P tile
    unsigned short* pw = psh[wave];

    const float sc = 0.08838834764831845f;  // 1/sqrt(128)
    const int rofs = (lane >> 4) << 3;
    const int kbB  = (lane >> 4) << 4;

    for (int kt = 0; kt < LSEQ / 32; ++kt) {
        v8f s0 = vzero, s1 = vzero;
        const int n0 = kt * 32 + (lane & 15);
#pragma unroll
        for (int j = 0; j < 4; ++j) {
            Frag16 b0, b1;
            const unsigned short* p0 = Kh + (size_t)n0 * HDIM + j * 32 + kbB;
            b0.q[0] = *(const U4*)p0;  b0.q[1] = *(const U4*)(p0 + 8);
            const unsigned short* p1 = Kh + (size_t)(n0 + 16) * HDIM + j * 32 + kbB;
            b1.q[0] = *(const U4*)p1;  b1.q[1] = *(const U4*)(p1 + 8);
            s0 = wmma_bf16(qf[j], b0, s0);
            s1 = wmma_bf16(qf[j], b1, s1);
        }
        // online softmax across the 32-key strip
#pragma unroll
        for (int v = 0; v < 8; ++v) {
            float a0 = s0[v] * sc, a1 = s1[v] * sc;
            float rm = fmaxf(a0, a1);
            for (int m = 1; m < 16; m <<= 1) rm = fmaxf(rm, __shfl_xor(rm, m));
            const float nm = fmaxf(mrow[v], rm);
            const float alpha = __expf(mrow[v] - nm);
            mrow[v] = nm;
            const float p0 = __expf(a0 - nm);
            const float p1 = __expf(a1 - nm);
            float rs = p0 + p1;
            for (int m = 1; m < 16; m <<= 1) rs += __shfl_xor(rs, m);
            lrow[v] = lrow[v] * alpha + rs;
#pragma unroll
            for (int dt = 0; dt < 8; ++dt) o[dt][v] = o[dt][v] * alpha;
            const int r = v + rofs;
            pw[r * 32 + (lane & 15)]      = f2bf(p0);
            pw[r * 32 + 16 + (lane & 15)] = f2bf(p1);
        }
        // P (C layout) -> A fragment via per-wave LDS tile
        Frag16 pa;
        {
            const unsigned short* pp = pw + (lane & 15) * 32 + ((lane >> 4) << 4);
            pa.q[0] = *(const U4*)pp;
            pa.q[1] = *(const U4*)(pp + 8);
        }
        // O += P @ V  (V^T gives contiguous K per lane)
#pragma unroll
        for (int dt = 0; dt < 8; ++dt) {
            Frag16 bv;
            const unsigned short* vp =
                Vh + (size_t)(dt * 16 + (lane & 15)) * LSEQ + kt * 32 + kbB;
            bv.q[0] = *(const U4*)vp;  bv.q[1] = *(const U4*)(vp + 8);
            o[dt] = wmma_bf16(pa, bv, o[dt]);
        }
    }
#pragma unroll
    for (int dt = 0; dt < 8; ++dt)
#pragma unroll
    for (int v = 0; v < 8; ++v) {
        const int row = qb * 16 + v + rofs;
        const int d = dt * 16 + (lane & 15);
        O[(size_t)row * HIDDEN + h * HDIM + d] = f2bf(o[dt][v] / lrow[v]);
    }
}

// ---------------------------------------------------------------------------
extern "C" void kernel_launch(void* const* d_in, const int* in_sizes, int n_in,
                              void* d_out, int out_size, void* d_ws, size_t ws_size,
                              hipStream_t stream) {
    (void)in_sizes; (void)n_in; (void)out_size; (void)ws_size;

    const float* img        = (const float*)d_in[0];
    const float* txt        = (const float*)d_in[1];
    const float* img_mod1   = (const float*)d_in[2];
    const float* img_mod2   = (const float*)d_in[3];
    const float* txt_mod1   = (const float*)d_in[4];
    const float* txt_mod2   = (const float*)d_in[5];
    const float* img_qkv_w  = (const float*)d_in[6];
    const float* img_qkv_b  = (const float*)d_in[7];
    const float* img_q_scale= (const float*)d_in[8];
    const float* img_k_scale= (const float*)d_in[9];
    const float* img_proj_w = (const float*)d_in[10];
    const float* img_proj_b = (const float*)d_in[11];
    const float* img_mlp_w1 = (const float*)d_in[12];
    const float* img_mlp_b1 = (const float*)d_in[13];
    const float* img_mlp_w2 = (const float*)d_in[14];
    const float* img_mlp_b2 = (const float*)d_in[15];
    const float* txt_qkv_w  = (const float*)d_in[16];
    const float* txt_qkv_b  = (const float*)d_in[17];
    const float* txt_q_scale= (const float*)d_in[18];
    const float* txt_k_scale= (const float*)d_in[19];
    const float* txt_proj_w = (const float*)d_in[20];
    const float* txt_proj_b = (const float*)d_in[21];
    const float* txt_mlp_w1 = (const float*)d_in[22];
    const float* txt_mlp_b1 = (const float*)d_in[23];
    const float* txt_mlp_w2 = (const float*)d_in[24];
    const float* txt_mlp_b2 = (const float*)d_in[25];
    const float* pe         = (const float*)d_in[26];

    char* ws = (char*)d_ws;
    size_t off = 0;
    auto take = [&](size_t bytes) -> char* {
        char* p = ws + off;
        off += (bytes + 255) & ~(size_t)255;
        return p;
    };
    unsigned short* xm_img  = (unsigned short*)take((size_t)TIMG * HIDDEN * 2);
    unsigned short* xm_txt  = (unsigned short*)take((size_t)TTXT * HIDDEN * 2);
    float*          qkv_img = (float*)take((size_t)TIMG * 3 * HIDDEN * 4);
    float*          qkv_txt = (float*)take((size_t)TTXT * 3 * HIDDEN * 4);
    unsigned short* Qb      = (unsigned short*)take((size_t)NHEAD * LSEQ * HDIM * 2);
    unsigned short* Kb      = (unsigned short*)take((size_t)NHEAD * LSEQ * HDIM * 2);
    unsigned short* Vt      = (unsigned short*)take((size_t)NHEAD * HDIM * LSEQ * 2);
    unsigned short* attn    = (unsigned short*)take((size_t)LSEQ * HIDDEN * 2);
    unsigned short* h_img   = (unsigned short*)take((size_t)TIMG * MLPD * 2);
    unsigned short* h_txt   = (unsigned short*)take((size_t)TTXT * MLPD * 2);

    float* out_img = (float*)d_out;
    float* out_txt = out_img + (size_t)TIMG * HIDDEN;

    // 1) LN + modulation (stage 1)
    ln_mod_kernel<<<TIMG, 256, 0, stream>>>(img, img_mod1, xm_img);
    ln_mod_kernel<<<TTXT, 256, 0, stream>>>(txt, txt_mod1, xm_txt);

    // 2) QKV GEMMs (f32 out)
    gemm_kernel<0><<<dim3(TIMG / 128, (3 * HIDDEN) / 256), 256, 0, stream>>>(
        xm_img, img_qkv_w, img_qkv_b, qkv_img, nullptr, nullptr, nullptr,
        TIMG, 3 * HIDDEN, HIDDEN);
    gemm_kernel<0><<<dim3(TTXT / 128, (3 * HIDDEN) / 256), 256, 0, stream>>>(
        xm_txt, txt_qkv_w, txt_qkv_b, qkv_txt, nullptr, nullptr, nullptr,
        TTXT, 3 * HIDDEN, HIDDEN);

    // 3) RMS + RoPE, pack Q/K/V^T (txt occupies s=0..511, img s=512..2047)
    qkv_post_kernel<<<dim3(TTXT, NHEAD), 128, 0, stream>>>(
        qkv_txt, txt_q_scale, txt_k_scale, pe, Qb, Kb, Vt, 0);
    qkv_post_kernel<<<dim3(TIMG, NHEAD), 128, 0, stream>>>(
        qkv_img, img_q_scale, img_k_scale, pe, Qb, Kb, Vt, TTXT);

    // 4) attention
    flash_kernel<<<dim3(LSEQ / 128, NHEAD), 256, 0, stream>>>(Qb, Kb, Vt, attn);

    // 5) proj + gated residual -> d_out
    gemm_kernel<2><<<dim3(TIMG / 128, HIDDEN / 256), 256, 0, stream>>>(
        attn + (size_t)TTXT * HIDDEN, img_proj_w, img_proj_b, out_img, nullptr,
        img, img_mod1 + 2 * HIDDEN, TIMG, HIDDEN, HIDDEN);
    gemm_kernel<2><<<dim3(TTXT / 128, HIDDEN / 256), 256, 0, stream>>>(
        attn, txt_proj_w, txt_proj_b, out_txt, nullptr,
        txt, txt_mod1 + 2 * HIDDEN, TTXT, HIDDEN, HIDDEN);

    // 6) LN + modulation (stage 2), reuse xm buffers
    ln_mod_kernel<<<TIMG, 256, 0, stream>>>(out_img, img_mod2, xm_img);
    ln_mod_kernel<<<TTXT, 256, 0, stream>>>(out_txt, txt_mod2, xm_txt);

    // 7) MLP up-proj + tanh-GELU (bf16 out)
    gemm_kernel<1><<<dim3(TIMG / 128, MLPD / 256), 256, 0, stream>>>(
        xm_img, img_mlp_w1, img_mlp_b1, nullptr, h_img, nullptr, nullptr,
        TIMG, MLPD, HIDDEN);
    gemm_kernel<1><<<dim3(TTXT / 128, MLPD / 256), 256, 0, stream>>>(
        xm_txt, txt_mlp_w1, txt_mlp_b1, nullptr, h_txt, nullptr, nullptr,
        TTXT, MLPD, HIDDEN);

    // 8) MLP down-proj + gated residual (txt also clipped to +-65504)
    gemm_kernel<2><<<dim3(TIMG / 128, HIDDEN / 256), 256, 0, stream>>>(
        h_img, img_mlp_w2, img_mlp_b2, out_img, nullptr,
        out_img, img_mod2 + 2 * HIDDEN, TIMG, HIDDEN, MLPD);
    gemm_kernel<3><<<dim3(TTXT / 128, HIDDEN / 256), 256, 0, stream>>>(
        h_txt, txt_mlp_w2, txt_mlp_b2, out_txt, nullptr,
        out_txt, txt_mod2 + 2 * HIDDEN, TTXT, HIDDEN, MLPD);
}
